// PolicyTransformer_86603720557329
// MI455X (gfx1250) — matrix operand order repaired and checked by reading, not compiled, not run on hardware
//
#include <hip/hip_runtime.h>
#include <math.h>

// ---------------- problem constants ----------------
#define S_LEN  1024
#define BATCH  2
#define TD_IN  138
#define TD_PAD 160          // 138 padded to multiple of 32 for WMMA K-loop
#define DMODEL 1024
#define NHEAD  16
#define DHEAD  64
#define NLAYER 8
#define FDIM   2048
#define MROWS  (S_LEN * BATCH)   // 2048 token rows

typedef __attribute__((ext_vector_type(16))) _Float16 v16h;
typedef __attribute__((ext_vector_type(8)))  _Float16 v8h;
typedef __attribute__((ext_vector_type(8)))  float    v8f;

static __device__ __forceinline__ v16h cat16(v8h a, v8h b) {
  return __builtin_shufflevector(a, b, 0,1,2,3,4,5,6,7,8,9,10,11,12,13,14,15);
}
static __device__ __forceinline__ v8f wmma_f16(v16h a, v16h b, v8f c) {
  // D(16x16 f32) = A(16x32 f16) * B(32x16 f16) + C
  return __builtin_amdgcn_wmma_f32_16x16x32_f16(false, a, false, b, (short)0, c, false, false);
}

// ---------------- conversions (weights and tokens, once per launch) ----------------
__global__ __launch_bounds__(256) void f32tof16_k(const float* __restrict__ in,
                                                  _Float16* __restrict__ out, int n) {
  int i = blockIdx.x * 256 + threadIdx.x;
  if (i < n) out[i] = (_Float16)in[i];
}

__global__ __launch_bounds__(256) void pad_convert_k(const float* __restrict__ in,
                                                     _Float16* __restrict__ out,
                                                     int rows, int K, int Kp) {
  int i = blockIdx.x * 256 + threadIdx.x;
  if (i >= rows * Kp) return;
  int k = i % Kp, r = i / Kp;
  out[i] = (k < K) ? (_Float16)in[(size_t)r * K + k] : (_Float16)0.0f;
}

// ---------------- WMMA GEMM: Y[M,N] = A[M,K] * B[N,K]^T + bias (+pos)(+res)(relu) --------
// Block tile 128x64, 8 waves, each wave owns a 32x32 output tile (2x2 WMMA accumulators).
// Double-buffered LDS (one barrier per 32-K step), next slab's global loads issued before
// the WMMAs, plus a 2-slab-ahead prefetch of both streams. Epilogue can emit f32 (C) and/or
// f16 (C16) so no separate conversion pass is ever needed.
__global__ __launch_bounds__(256) void gemm_wmma_k(
    const _Float16* __restrict__ A, const _Float16* __restrict__ Bw,
    const float* __restrict__ bias, const float* __restrict__ residual,
    const float* __restrict__ pos, float* __restrict__ C, _Float16* __restrict__ C16,
    int M, int N, int K, int relu)
{
  __shared__ _Float16 As[2][128][40];  // 128x32 tile, stride 40 (80B rows, 16B aligned)
  __shared__ _Float16 Bs[2][64][40];   //  64x32 tile

  const int tid  = threadIdx.x;
  const int lane = tid & 31;
  const int wave = tid >> 5;          // 8 waves
  const int wm   = wave & 3;          // 4 M-tiles of 32
  const int wn   = wave >> 2;         // 2 N-tiles of 32
  const int m0   = blockIdx.y * 128;
  const int n0   = blockIdx.x * 64;

  const int arow   = tid >> 2;        // 0..63 (A covers rows arow, arow+64)
  const int achunk = (tid & 3) * 8;   // 8-half chunk within 32-K slab

  const int l15     = lane & 15;
  const int kbA     = (lane < 16) ? 0 : 8;    // A-frag K sub-offset (ISA 16-bit A layout)
  const int kbB     = (lane < 16) ? 0 : 8;    // B-frag K sub-offset (ISA 16-bit B layout)
  const int halfOff = (lane >> 4) * 8;        // C-frag row offset

  const _Float16* aP0 = A  + (size_t)(m0 + arow)      * K + achunk;
  const _Float16* aP1 = A  + (size_t)(m0 + arow + 64) * K + achunk;
  const _Float16* bP0 = Bw + (size_t)(n0 + arow)      * K + achunk;

  // preload k-slab 0 into buffer 0
  {
    v8h a0 = *(const v8h*)aP0;
    v8h a1 = *(const v8h*)aP1;
    v8h b0 = *(const v8h*)bP0;
    *(v8h*)&As[0][arow][achunk]      = a0;
    *(v8h*)&As[0][arow + 64][achunk] = a1;
    *(v8h*)&Bs[0][arow][achunk]      = b0;
  }
  __syncthreads();

  v8f acc00 = {}, acc01 = {}, acc10 = {}, acc11 = {};
  const int mr0 = wm * 32 + l15, mr1 = mr0 + 16;
  const int nc0 = wn * 32 + l15, nc1 = nc0 + 16;

  int p = 0;
  for (int k0 = 0; k0 < K; k0 += 32) {
    const int kn = k0 + 32;
    v8h na0, na1, nb0;
    if (kn < K) {                       // issue next slab's global loads first
      na0 = *(const v8h*)(aP0 + kn);
      na1 = *(const v8h*)(aP1 + kn);
      nb0 = *(const v8h*)(bP0 + kn);
    }
    if (kn + 32 < K) {                  // prefetch 2 slabs ahead (global_prefetch_b8)
      __builtin_prefetch(aP0 + kn + 32, 0, 0);
      __builtin_prefetch(bP0 + kn + 32, 0, 0);
    }

    v16h af0 = cat16(*(const v8h*)&As[p][mr0][kbA], *(const v8h*)&As[p][mr0][kbA + 16]);
    v16h af1 = cat16(*(const v8h*)&As[p][mr1][kbA], *(const v8h*)&As[p][mr1][kbA + 16]);
    v16h bf0 = cat16(*(const v8h*)&Bs[p][nc0][kbB], *(const v8h*)&Bs[p][nc0][kbB + 16]);
    v16h bf1 = cat16(*(const v8h*)&Bs[p][nc1][kbB], *(const v8h*)&Bs[p][nc1][kbB + 16]);

    acc00 = wmma_f16(af0, bf0, acc00);
    acc01 = wmma_f16(af0, bf1, acc01);
    acc10 = wmma_f16(af1, bf0, acc10);
    acc11 = wmma_f16(af1, bf1, acc11);

    if (kn < K) {                       // stage into the other buffer, single barrier
      *(v8h*)&As[p ^ 1][arow][achunk]      = na0;
      *(v8h*)&As[p ^ 1][arow + 64][achunk] = na1;
      *(v8h*)&Bs[p ^ 1][arow][achunk]      = nb0;
      __syncthreads();
      p ^= 1;
    }
  }

  #pragma unroll
  for (int mi = 0; mi < 2; ++mi) {
    #pragma unroll
    for (int r = 0; r < 8; ++r) {
      int m = m0 + wm * 32 + mi * 16 + r + halfOff;
      #pragma unroll
      for (int ni = 0; ni < 2; ++ni) {
        int n = n0 + wn * 32 + ni * 16 + l15;
        float v;
        if (mi == 0) v = (ni == 0 ? acc00[r] : acc01[r]);
        else         v = (ni == 0 ? acc10[r] : acc11[r]);
        v += bias[n];
        if (pos)      v += pos[(size_t)(m / BATCH) * N + n];     // pos_emb[s][n]
        if (residual) v += residual[(size_t)m * N + n];
        if (relu)     v = fmaxf(v, 0.0f);
        if (C)   C[(size_t)m * N + n]   = v;
        if (C16) C16[(size_t)m * N + n] = (_Float16)v;
      }
    }
  }
}

// ---------------- RoPE + head pack: (S,B,D) f32 -> (B,H,S,Dh) f16 ----------------
__global__ __launch_bounds__(256) void rope_pack_k(const float* __restrict__ in,
                                                   _Float16* __restrict__ out) {
  int idx = blockIdx.x * 256 + threadIdx.x;          // over S*B*D/2 pairs
  if (idx >= S_LEN * BATCH * (DMODEL / 2)) return;
  int d2 = idx & (DMODEL / 2 - 1);
  int sb = idx >> 9;                                  // / (D/2)
  int b  = sb & (BATCH - 1);
  int s  = sb >> 1;
  int d   = d2 * 2;
  int h   = d >> 6;
  int dhi = d & 63;
  int j   = dhi >> 1;
  float inv = __powf(10000.0f, -(float)(2 * j) / 64.0f);
  float c, sn;
  __sincosf((float)s * inv, &c, &sn);
  size_t src = (size_t)(s * BATCH + b) * DMODEL + d;
  float x0 = in[src], x1 = in[src + 1];
  size_t dst = ((size_t)(b * NHEAD + h) * S_LEN + s) * DHEAD + dhi;
  out[dst]     = (_Float16)(x0 * c - x1 * sn);
  out[dst + 1] = (_Float16)(x1 * c + x0 * sn);
}

// ---------------- V pack transposed: (S,B,D) f32 -> (B,H,Dh,S) f16 ----------------
__global__ __launch_bounds__(256) void v_pack_k(const float* __restrict__ in,
                                                _Float16* __restrict__ out) {
  int idx = blockIdx.x * 256 + threadIdx.x;          // over S*B*D
  if (idx >= S_LEN * BATCH * DMODEL) return;
  int d  = idx & (DMODEL - 1);
  int sb = idx >> 10;
  int b  = sb & (BATCH - 1);
  int s  = sb >> 1;
  int h   = d >> 6;
  int dhi = d & 63;
  out[((size_t)(b * NHEAD + h) * DHEAD + dhi) * S_LEN + s] = (_Float16)in[idx];
}

// ---------------- flash attention: per-wave 16-row query tile, 32-key steps ----------------
// Q,K: (B,H,S,Dh) f16; Vt: (B,H,Dh,S) f16; O16: (S,B,D) f16 (consumed by o-proj GEMM)
__global__ __launch_bounds__(256) void flash_attn_k(
    const _Float16* __restrict__ Q, const _Float16* __restrict__ Kx,
    const _Float16* __restrict__ Vt, _Float16* __restrict__ O16)
{
  __shared__ _Float16 Plds[8][16][40];   // per-wave P tile (16x32), padded
  const int lane = threadIdx.x & 31;
  const int wave = threadIdx.x >> 5;
  const int q0 = blockIdx.x * 128 + wave * 16;
  const int bh = blockIdx.y;
  const int b  = bh >> 4;
  const int h  = bh & 15;
  const int l15     = lane & 15;
  const int kbA     = (lane < 16) ? 0 : 8;
  const int kbB     = (lane < 16) ? 0 : 8;
  const int halfOff = (lane >> 4) * 8;

  // Q fragments covering Dh 0..31 and 32..63
  const _Float16* qp = Q + ((size_t)bh * S_LEN + (q0 + l15)) * DHEAD;
  v16h qA0 = cat16(*(const v8h*)(qp + kbA),      *(const v8h*)(qp + kbA + 16));
  v16h qA1 = cat16(*(const v8h*)(qp + 32 + kbA), *(const v8h*)(qp + 32 + kbA + 16));

  float mr[8], lr[8];
  #pragma unroll
  for (int r = 0; r < 8; ++r) { mr[r] = -3.0e38f; lr[r] = 0.0f; }
  v8f o0 = {}, o1 = {}, o2 = {}, o3 = {};

  const float scale = 0.125f;   // 1/sqrt(64)

  for (int k0 = 0; k0 < q0 + 16; k0 += 32) {
    // ---- scores S = Q K^T for 16 rows x 32 keys (two 16x16 tiles) ----
    v8f s0 = {}, s1 = {};
    {
      const _Float16* kp0 = Kx + ((size_t)bh * S_LEN + (k0 + l15)) * DHEAD;
      v16h kf00 = cat16(*(const v8h*)(kp0 + kbB),      *(const v8h*)(kp0 + kbB + 16));
      v16h kf01 = cat16(*(const v8h*)(kp0 + 32 + kbB), *(const v8h*)(kp0 + 32 + kbB + 16));
      s0 = wmma_f16(qA0, kf00, s0);
      s0 = wmma_f16(qA1, kf01, s0);
      const _Float16* kp1 = kp0 + (size_t)16 * DHEAD;
      v16h kf10 = cat16(*(const v8h*)(kp1 + kbB),      *(const v8h*)(kp1 + kbB + 16));
      v16h kf11 = cat16(*(const v8h*)(kp1 + 32 + kbB), *(const v8h*)(kp1 + 32 + kbB + 16));
      s1 = wmma_f16(qA0, kf10, s1);
      s1 = wmma_f16(qA1, kf11, s1);
    }

    // ---- causal mask + online softmax (row = r + halfOff, col = lane&15) ----
    #pragma unroll
    for (int r = 0; r < 8; ++r) {
      int rowq = q0 + r + halfOff;
      int c0 = k0 + l15, c1 = c0 + 16;
      float x0 = (c0 <= rowq) ? s0[r] * scale : -3.0e38f;
      float x1 = (c1 <= rowq) ? s1[r] * scale : -3.0e38f;
      float mx = fmaxf(x0, x1);
      #pragma unroll
      for (int i = 1; i < 16; i <<= 1) mx = fmaxf(mx, __shfl_xor(mx, i, 32));
      float mnew  = fmaxf(mr[r], mx);
      float alpha = __expf(mr[r] - mnew);
      float p0 = __expf(x0 - mnew);
      float p1 = __expf(x1 - mnew);
      float rs = p0 + p1;
      #pragma unroll
      for (int i = 1; i < 16; i <<= 1) rs += __shfl_xor(rs, i, 32);
      lr[r] = lr[r] * alpha + rs;
      mr[r] = mnew;
      o0[r] *= alpha; o1[r] *= alpha; o2[r] *= alpha; o3[r] *= alpha;
      Plds[wave][r + halfOff][l15]      = (_Float16)p0;
      Plds[wave][r + halfOff][16 + l15] = (_Float16)p1;
    }
    asm volatile("s_wait_dscnt 0x0" ::: "memory");

    // repack P (16x32) into WMMA A-fragment layout
    v16h pA = cat16(*(const v8h*)&Plds[wave][l15][kbA],
                    *(const v8h*)&Plds[wave][l15][kbA + 16]);

    // ---- O += P * V (Vt gives contiguous key runs per dh row) ----
    const _Float16* vp = Vt + (size_t)bh * DHEAD * S_LEN + k0 + kbB;
    {
      const _Float16* v0 = vp + (size_t)(0 * 16 + l15) * S_LEN;
      o0 = wmma_f16(pA, cat16(*(const v8h*)v0, *(const v8h*)(v0 + 16)), o0);
      const _Float16* v1 = vp + (size_t)(1 * 16 + l15) * S_LEN;
      o1 = wmma_f16(pA, cat16(*(const v8h*)v1, *(const v8h*)(v1 + 16)), o1);
      const _Float16* v2 = vp + (size_t)(2 * 16 + l15) * S_LEN;
      o2 = wmma_f16(pA, cat16(*(const v8h*)v2, *(const v8h*)(v2 + 16)), o2);
      const _Float16* v3 = vp + (size_t)(3 * 16 + l15) * S_LEN;
      o3 = wmma_f16(pA, cat16(*(const v8h*)v3, *(const v8h*)(v3 + 16)), o3);
    }
  }

  // ---- normalize and write O in (S,B,D) f16 ----
  #pragma unroll
  for (int r = 0; r < 8; ++r) {
    int rowq = q0 + r + halfOff;
    float inv = 1.0f / lr[r];
    size_t base = ((size_t)rowq * BATCH + b) * DMODEL + (size_t)h * DHEAD;
    O16[base + 0 * 16 + l15] = (_Float16)(o0[r] * inv);
    O16[base + 1 * 16 + l15] = (_Float16)(o1[r] * inv);
    O16[base + 2 * 16 + l15] = (_Float16)(o2[r] * inv);
    O16[base + 3 * 16 + l15] = (_Float16)(o3[r] * inv);
  }
}

// ---------------- LayerNorm over D=1024, one block per row; dual f32+f16 output ----------
__global__ __launch_bounds__(256) void layernorm_k(const float* __restrict__ in,
                                                   const float* __restrict__ g,
                                                   const float* __restrict__ be,
                                                   float* __restrict__ out,
                                                   _Float16* __restrict__ out16) {
  __shared__ float red[256];
  int m = blockIdx.x, t = threadIdx.x;
  const float* row = in + (size_t)m * DMODEL;
  float s = 0.0f;
  for (int i = t; i < DMODEL; i += 256) s += row[i];
  red[t] = s; __syncthreads();
  for (int o = 128; o > 0; o >>= 1) { if (t < o) red[t] += red[t + o]; __syncthreads(); }
  float mu = red[0] * (1.0f / DMODEL);
  __syncthreads();
  float v = 0.0f;
  for (int i = t; i < DMODEL; i += 256) { float d = row[i] - mu; v += d * d; }
  red[t] = v; __syncthreads();
  for (int o = 128; o > 0; o >>= 1) { if (t < o) red[t] += red[t + o]; __syncthreads(); }
  float rstd = rsqrtf(red[0] * (1.0f / DMODEL) + 1e-5f);
  float* orow = out + (size_t)m * DMODEL;
  _Float16* orow16 = out16 + (size_t)m * DMODEL;
  for (int i = t; i < DMODEL; i += 256) {
    float y = (row[i] - mu) * rstd * g[i] + be[i];
    orow[i]   = y;
    orow16[i] = (_Float16)y;
  }
}

// ---------------- head: out[m] = dot(x[m,:], Wh) + bh ----------------
__global__ __launch_bounds__(256) void head_k(const float* __restrict__ x,
                                              const float* __restrict__ Wh,
                                              const float* __restrict__ bh,
                                              float* __restrict__ out) {
  __shared__ float red[256];
  int m = blockIdx.x, t = threadIdx.x;
  const float* row = x + (size_t)m * DMODEL;
  float s = 0.0f;
  for (int i = t; i < DMODEL; i += 256) s += row[i] * Wh[i];
  red[t] = s; __syncthreads();
  for (int o = 128; o > 0; o >>= 1) { if (t < o) red[t] += red[t + o]; __syncthreads(); }
  if (t == 0) out[m] = red[0] + bh[0];
}

// ---------------- launcher ----------------
extern "C" void kernel_launch(void* const* d_in, const int* in_sizes, int n_in,
                              void* d_out, int out_size, void* d_ws, size_t ws_size,
                              hipStream_t stream) {
  (void)in_sizes; (void)n_in; (void)out_size; (void)ws_size;
  const float* tokens = (const float*)d_in[0];
  const float* Win    = (const float*)d_in[1];
  const float* bin_   = (const float*)d_in[2];
  const float* pos    = (const float*)d_in[3];
  const float* Wq = (const float*)d_in[4];   const float* bq = (const float*)d_in[5];
  const float* Wk = (const float*)d_in[6];   const float* bk = (const float*)d_in[7];
  const float* Wv = (const float*)d_in[8];   const float* bv = (const float*)d_in[9];
  const float* Wo = (const float*)d_in[10];  const float* bo = (const float*)d_in[11];
  const float* W1 = (const float*)d_in[12];  const float* b1 = (const float*)d_in[13];
  const float* W2 = (const float*)d_in[14];  const float* b2 = (const float*)d_in[15];
  const float* g1 = (const float*)d_in[16];  const float* be1 = (const float*)d_in[17];
  const float* g2 = (const float*)d_in[18];  const float* be2 = (const float*)d_in[19];
  const float* Wh = (const float*)d_in[20];  const float* bhp = (const float*)d_in[21];
  float* out = (float*)d_out;

  char* cur = (char*)d_ws;
  auto alloc = [&](size_t bytes) -> void* {
    void* r = (void*)cur;
    cur += (bytes + 255) & ~(size_t)255;
    return r;
  };

  _Float16* tok16 = (_Float16*)alloc((size_t)MROWS * TD_PAD * 2);
  _Float16* win16 = (_Float16*)alloc((size_t)DMODEL * TD_PAD * 2);
  _Float16* wq16  = (_Float16*)alloc((size_t)NLAYER * DMODEL * DMODEL * 2);
  _Float16* wk16  = (_Float16*)alloc((size_t)NLAYER * DMODEL * DMODEL * 2);
  _Float16* wv16  = (_Float16*)alloc((size_t)NLAYER * DMODEL * DMODEL * 2);
  _Float16* wo16  = (_Float16*)alloc((size_t)NLAYER * DMODEL * DMODEL * 2);
  _Float16* w116  = (_Float16*)alloc((size_t)NLAYER * FDIM * DMODEL * 2);
  _Float16* w216  = (_Float16*)alloc((size_t)NLAYER * DMODEL * FDIM * 2);
  float*    x     = (float*)   alloc((size_t)MROWS * DMODEL * 4);
  float*    tmp   = (float*)   alloc((size_t)MROWS * DMODEL * 4);
  float*    qb    = (float*)   alloc((size_t)MROWS * DMODEL * 4);
  float*    kb    = (float*)   alloc((size_t)MROWS * DMODEL * 4);
  float*    vb    = (float*)   alloc((size_t)MROWS * DMODEL * 4);
  _Float16* x16   = (_Float16*)alloc((size_t)MROWS * DMODEL * 2);
  _Float16* q16   = (_Float16*)alloc((size_t)MROWS * DMODEL * 2);
  _Float16* k16   = (_Float16*)alloc((size_t)MROWS * DMODEL * 2);
  _Float16* vt16  = (_Float16*)alloc((size_t)MROWS * DMODEL * 2);
  _Float16* o16   = (_Float16*)alloc((size_t)MROWS * DMODEL * 2);
  _Float16* h16   = (_Float16*)alloc((size_t)MROWS * FDIM * 2);

  // weight / input conversion to f16 (deterministic each launch)
  {
    int n = MROWS * TD_PAD;
    pad_convert_k<<<(n + 255) / 256, 256, 0, stream>>>(tokens, tok16, MROWS, TD_IN, TD_PAD);
    n = DMODEL * TD_PAD;
    pad_convert_k<<<(n + 255) / 256, 256, 0, stream>>>(Win, win16, DMODEL, TD_IN, TD_PAD);
    int nw = NLAYER * DMODEL * DMODEL;
    f32tof16_k<<<(nw + 255) / 256, 256, 0, stream>>>(Wq, wq16, nw);
    f32tof16_k<<<(nw + 255) / 256, 256, 0, stream>>>(Wk, wk16, nw);
    f32tof16_k<<<(nw + 255) / 256, 256, 0, stream>>>(Wv, wv16, nw);
    f32tof16_k<<<(nw + 255) / 256, 256, 0, stream>>>(Wo, wo16, nw);
    int nf = NLAYER * FDIM * DMODEL;
    f32tof16_k<<<(nf + 255) / 256, 256, 0, stream>>>(W1, w116, nf);
    f32tof16_k<<<(nf + 255) / 256, 256, 0, stream>>>(W2, w216, nf);
  }

  dim3 blk(256);
  dim3 gD(DMODEL / 64, MROWS / 128);   // 16 x 16 blocks
  dim3 gF(FDIM / 64, MROWS / 128);     // 32 x 16 blocks
  const int nact = MROWS * DMODEL;
  const int nro  = MROWS * DMODEL / 2;

  // input projection: x(+x16) = tokens @ Win^T + bin + pos_emb
  gemm_wmma_k<<<gD, blk, 0, stream>>>(tok16, win16, bin_, nullptr, pos, x, x16,
                                      MROWS, DMODEL, TD_PAD, 0);

  for (int l = 0; l < NLAYER; ++l) {
    const _Float16* wq_l = wq16 + (size_t)l * DMODEL * DMODEL;
    const _Float16* wk_l = wk16 + (size_t)l * DMODEL * DMODEL;
    const _Float16* wv_l = wv16 + (size_t)l * DMODEL * DMODEL;
    const _Float16* wo_l = wo16 + (size_t)l * DMODEL * DMODEL;
    const _Float16* w1_l = w116 + (size_t)l * FDIM * DMODEL;
    const _Float16* w2_l = w216 + (size_t)l * DMODEL * FDIM;

    // Q/K/V projections (f32 out, consumed by RoPE / V-pack in full precision)
    gemm_wmma_k<<<gD, blk, 0, stream>>>(x16, wq_l, bq + l * DMODEL, nullptr, nullptr,
                                        qb, nullptr, MROWS, DMODEL, DMODEL, 0);
    gemm_wmma_k<<<gD, blk, 0, stream>>>(x16, wk_l, bk + l * DMODEL, nullptr, nullptr,
                                        kb, nullptr, MROWS, DMODEL, DMODEL, 0);
    gemm_wmma_k<<<gD, blk, 0, stream>>>(x16, wv_l, bv + l * DMODEL, nullptr, nullptr,
                                        vb, nullptr, MROWS, DMODEL, DMODEL, 0);

    rope_pack_k<<<(nro + 255) / 256, blk, 0, stream>>>(qb, q16);
    rope_pack_k<<<(nro + 255) / 256, blk, 0, stream>>>(kb, k16);
    v_pack_k<<<(nact + 255) / 256, blk, 0, stream>>>(vb, vt16);

    dim3 gA(S_LEN / 128, BATCH * NHEAD);
    flash_attn_k<<<gA, blk, 0, stream>>>(q16, k16, vt16, o16);   // f16 out, no extra pass

    // o-projection + residual(x) -> tmp ; LN -> x (f32) + x16 (f16)
    gemm_wmma_k<<<gD, blk, 0, stream>>>(o16, wo_l, bo + l * DMODEL, x, nullptr,
                                        tmp, nullptr, MROWS, DMODEL, DMODEL, 0);
    layernorm_k<<<MROWS, blk, 0, stream>>>(tmp, g1 + l * DMODEL, be1 + l * DMODEL, x, x16);

    // FFN1 (ReLU) emits f16 only — f32 copy is never consumed
    gemm_wmma_k<<<gF, blk, 0, stream>>>(x16, w1_l, b1 + l * FDIM, nullptr, nullptr,
                                        nullptr, h16, MROWS, FDIM, DMODEL, 1);
    // FFN2 + residual(x) -> tmp ; LN -> x + x16
    gemm_wmma_k<<<gD, blk, 0, stream>>>(h16, w2_l, b2 + l * DMODEL, x, nullptr,
                                        tmp, nullptr, MROWS, DMODEL, FDIM, 0);
    layernorm_k<<<MROWS, blk, 0, stream>>>(tmp, g2 + l * DMODEL, be2 + l * DMODEL, x, x16);
  }

  head_k<<<MROWS, blk, 0, stream>>>(x, Wh, bhp, out);
}
